// MultiHeadAttention_69681549410872
// MI455X (gfx1250) — matrix operand order repaired and checked by reading, not compiled, not run on hardware
//
#include <hip/hip_runtime.h>

// ---------------------------------------------------------------------------
// MI455X (gfx1250) fused multi-head attention (no softmax), bf16 WMMA pipeline
//   B=2, L=2048, D_MODEL=1024, H=16, Dh=64, HD=1024
//   - QKV/out projections: per-wave 16x64 WMMA tiles, bf16, f32 accumulate
//   - attention: per-block LDS staging of K/V via async global->LDS DMA
//     (ASYNCcnt), double buffered; S^T accumulator reused in-register as the
//     bf16 A operand of the S@V WMMA (no cross-lane repack).
// ---------------------------------------------------------------------------

typedef __bf16 bf16_t;
typedef __attribute__((ext_vector_type(16))) __bf16 v16bf;
typedef __attribute__((ext_vector_type(8)))  __bf16 v8bf;
typedef __attribute__((ext_vector_type(8)))  float  v8f;
typedef __attribute__((ext_vector_type(4)))  int    v4i;

#define LSEQ   2048
#define DMODEL 1024
#define NHEAD  16
#define HDIM   64
#define BL     4096              // B * L
#define HD     1024              // NHEAD * HDIM

// ---- async global->LDS copy (16 bytes), with portable fallback ------------
#if defined(__has_builtin)
#  if __has_builtin(__builtin_amdgcn_global_load_async_to_lds_b128)
#    define HAVE_ASYNC_LDS 1
#  endif
#endif
#ifndef HAVE_ASYNC_LDS
#  define HAVE_ASYNC_LDS 0
#endif

__device__ __forceinline__ void async_copy16(const bf16_t* __restrict__ g,
                                             bf16_t* __restrict__ l) {
#if HAVE_ASYNC_LDS
    __builtin_amdgcn_global_load_async_to_lds_b128(
        (__attribute__((address_space(1))) v4i*)g,
        (__attribute__((address_space(3))) v4i*)l,
        /*imm offset=*/0, /*cpol=*/0);
#else
    *(v8bf*)l = *(const v8bf*)g;           // sync fallback: global->VGPR->LDS
#endif
}

__device__ __forceinline__ void wait_async_lds() {
#if HAVE_ASYNC_LDS
    asm volatile("s_wait_asynccnt 0x0" ::: "memory");
#endif
}

// ---- WMMA wrapper: D = A(16x32 bf16) * B(32x16 bf16) + C(16x16 f32) --------
__device__ __forceinline__ v8f wmma_bf16(v16bf a, v16bf b, v8f c) {
    return __builtin_amdgcn_wmma_f32_16x16x32_bf16(
        /*neg_a=*/false, a, /*neg_b=*/false, b,
        /*c_mod=*/(short)0, c, /*reuse_a=*/false, /*reuse_b=*/false);
}

// ---- A operand (16x32 bf16) from row-major [rows x ld] --------------------
// ISA layout: lanes 0-15 row M=lane, halves = K {0..7, 16..23};
//             lanes 16-31 row M=lane-16, halves = K {8..15, 24..31}.
__device__ __forceinline__ v16bf frag_a_row(const bf16_t* __restrict__ base,
                                            int ld, int koff, int lane) {
    const int m  = lane & 15;
    const int kh = (lane >> 4) << 3;            // 0 or 8
    const bf16_t* p = base + m * ld + koff + kh;
    v8bf lo = *(const v8bf*)(p);                // K = koff+kh .. +7
    v8bf hi = *(const v8bf*)(p + 16);           // K = koff+kh+16 .. +23
    v16bf f;
#pragma unroll
    for (int i = 0; i < 8; ++i) { f[i] = lo[i]; f[8 + i] = hi[i]; }
    return f;
}

// ---- B operand (32x16 bf16) from "column rows": src[n][k] row-major -------
// ISA layout: lanes 0-15 col N=lane, K {0..7, 16..23}; lanes 16-31 col N=lane-16, K {8..15, 24..31}.
__device__ __forceinline__ v16bf frag_b_colrows(const bf16_t* __restrict__ base,
                                                int ld, int koff, int lane) {
    const int n  = lane & 15;
    const int kh = (lane >> 4) << 3;            // 0 or 8
    const bf16_t* p = base + n * ld + koff + kh;
    v8bf lo = *(const v8bf*)(p);
    v8bf hi = *(const v8bf*)(p + 16);
    v16bf f;
#pragma unroll
    for (int i = 0; i < 8; ++i) { f[i] = lo[i]; f[8 + i] = hi[i]; }
    return f;
}

// ---------------------------------------------------------------------------
// Phase 0a: f32 -> bf16 elementwise
// ---------------------------------------------------------------------------
__global__ void __launch_bounds__(256) cvt_bf16_kernel(const float* __restrict__ in,
                                                       bf16_t* __restrict__ out, int n) {
    int i = blockIdx.x * blockDim.x + threadIdx.x;
    if (i < n) out[i] = (bf16_t)in[i];
}

// ---------------------------------------------------------------------------
// Phase 0b: f32 [K x N] -> bf16 transposed [N x K] via LDS 32x32 tile
// ---------------------------------------------------------------------------
__global__ void __launch_bounds__(256) transpose_cvt_kernel(const float* __restrict__ in,
                                                            bf16_t* __restrict__ out) {
    __shared__ bf16_t tile[32][33];
    const int k0 = blockIdx.y * 32;
    const int n0 = blockIdx.x * 32;
#pragma unroll
    for (int i = threadIdx.y; i < 32; i += 8)
        tile[i][threadIdx.x] = (bf16_t)in[(k0 + i) * DMODEL + n0 + threadIdx.x];
    __syncthreads();
#pragma unroll
    for (int i = threadIdx.y; i < 32; i += 8)
        out[(size_t)(n0 + i) * DMODEL + k0 + threadIdx.x] = tile[threadIdx.x][i];
}

// ---------------------------------------------------------------------------
// Phase 1: projection GEMM.  dst = xb[BL x 1024] * W_t^T + bias
//   One wave computes a 16(M) x 64(N) tile; K loop of 32 with A-frag reuse.
//   vmode==0: write row-major per head  [b][h][l][d]   (Q, K)
//   vmode==1: write transposed per head [b][h][d][l]   (V)
// ---------------------------------------------------------------------------
__global__ void __launch_bounds__(256) proj_gemm_kernel(const bf16_t* __restrict__ xb,
                                                        const bf16_t* __restrict__ wt,
                                                        const float*  __restrict__ bias,
                                                        bf16_t* __restrict__ dst, int vmode) {
    const int lane = threadIdx.x & 31;
    const int w    = (blockIdx.x * blockDim.x + threadIdx.x) >> 5;
    const int mt   = w >> 4;                    // 0..255 : 16-row tiles over BL
    const int nt   = w & 15;                    // 0..15  : 64-col tiles over HD
    const bf16_t* abase = xb + (size_t)mt * 16 * DMODEL;

    v8f acc[4] = {};
    for (int k = 0; k < DMODEL; k += 32) {
        __builtin_prefetch(abase + ((lane & 15) * DMODEL) + k + 128, 0, 0);
        v16bf a = frag_a_row(abase, DMODEL, k, lane);
#pragma unroll
        for (int dt = 0; dt < 4; ++dt) {
            v16bf b = frag_b_colrows(wt + (size_t)(nt * 64 + dt * 16) * DMODEL, DMODEL, k, lane);
            acc[dt] = wmma_bf16(a, b, acc[dt]);
        }
    }
#pragma unroll
    for (int dt = 0; dt < 4; ++dt) {
        const int col = nt * 64 + dt * 16 + (lane & 15);
        const float bv = bias[col];
        const int h = col >> 6, d = col & 63;
#pragma unroll
        for (int r = 0; r < 8; ++r) {
            const int row = mt * 16 + r + ((lane >> 4) << 3);
            const int b = row >> 11, l = row & (LSEQ - 1);
            const float v = acc[dt][r] + bv;
            if (vmode) dst[((size_t)(b * NHEAD + h) * HDIM + d) * LSEQ + l] = (bf16_t)v;
            else       dst[((size_t)(b * NHEAD + h) * LSEQ + l) * HDIM + d] = (bf16_t)v;
        }
    }
}

// ---------------------------------------------------------------------------
// Phase 2: fused unnormalized attention.
//   Block = 8 waves, all on the same (b,h); waves own adjacent 16-row l-tiles.
//   K/V chunks (64 m-rows) staged in LDS via async global->LDS, double
//   buffered: DMA of chunk i+1 overlaps the 16 WMMAs of chunk i.
// ---------------------------------------------------------------------------
#define MCHUNK 64

__device__ __forceinline__ void stage_kv(const bf16_t* __restrict__ Kg,
                                         const bf16_t* __restrict__ Vtg,
                                         bf16_t* __restrict__ sk,
                                         bf16_t* __restrict__ sv,
                                         int m, int t) {
    // 256 threads move 8KB K + 8KB V: each thread 2x16B from K, 2x16B from V.
    const int r = t >> 2;              // 0..63  (K row m+r / V d-row r)
    const int c = (t & 3) << 4;        // 0,16,32,48 (elements)
    async_copy16(Kg + (size_t)(m + r) * HDIM + c,     sk + r * MCHUNK + c);
    async_copy16(Kg + (size_t)(m + r) * HDIM + c + 8, sk + r * MCHUNK + c + 8);
    async_copy16(Vtg + (size_t)r * LSEQ + m + c,      sv + r * MCHUNK + c);
    async_copy16(Vtg + (size_t)r * LSEQ + m + c + 8,  sv + r * MCHUNK + c + 8);
}

__global__ void __launch_bounds__(256) attn_kernel(const bf16_t* __restrict__ Q,
                                                   const bf16_t* __restrict__ K,
                                                   const bf16_t* __restrict__ Vt,
                                                   bf16_t* __restrict__ attn) {
    __shared__ bf16_t sk[2][MCHUNK * HDIM];     // K rows (m-major)   : 2 x 8KB
    __shared__ bf16_t sv[2][HDIM * MCHUNK];     // V^T rows (d-major) : 2 x 8KB

    const int t    = threadIdx.x;
    const int lane = t & 31;
    const int bh   = blockIdx.x >> 4;                       // 0..31
    const int lt   = ((blockIdx.x & 15) << 3) + (t >> 5);   // 0..127
    const float scale = 0.125f;                             // 1/sqrt(64)

    const bf16_t* Kg  = K  + (size_t)bh * LSEQ * HDIM;
    const bf16_t* Vtg = Vt + (size_t)bh * HDIM * LSEQ;

    // Per-wave Q fragments (B operand, cols = l), invariant over the m loop.
    const bf16_t* qbase = Q + ((size_t)bh * LSEQ + lt * 16) * HDIM;
    const v16bf qf0 = frag_b_colrows(qbase, HDIM, 0,  lane);
    const v16bf qf1 = frag_b_colrows(qbase, HDIM, 32, lane);

    stage_kv(Kg, Vtg, sk[0], sv[0], 0, t);                  // prologue DMA

    v8f acc[4] = {};
    for (int ms = 0; ms < LSEQ; ms += MCHUNK) {
        const int cur = (ms >> 6) & 1;
        wait_async_lds();                                   // this wave's DMA done
        __syncthreads();                                    // whole block's DMA done
        if (ms + MCHUNK < LSEQ)
            stage_kv(Kg, Vtg, sk[cur ^ 1], sv[cur ^ 1], ms + MCHUNK, t);

        const bf16_t* kbuf = sk[cur];
        const bf16_t* vbuf = sv[cur];
#pragma unroll
        for (int sub = 0; sub < MCHUNK; sub += 32) {
            // S^T tiles: D = K_rows x Q_cols  (f32 accum)
            const bf16_t* kb0 = kbuf + sub * HDIM;
            const bf16_t* kb1 = kb0 + 16 * HDIM;
            v8f s0 = {}, s1 = {};
            s0 = wmma_bf16(frag_a_row(kb0, HDIM, 0,  lane), qf0, s0);
            s0 = wmma_bf16(frag_a_row(kb0, HDIM, 32, lane), qf1, s0);
            s1 = wmma_bf16(frag_a_row(kb1, HDIM, 0,  lane), qf0, s1);
            s1 = wmma_bf16(frag_a_row(kb1, HDIM, 32, lane), qf1, s1);

            // D(S^T) -> bf16 A operand of (S @ V): per-lane scale+convert only.
            v16bf sa;
#pragma unroll
            for (int i = 0; i < 8; ++i) {
                sa[i]     = (bf16_t)(s0[i] * scale);
                sa[8 + i] = (bf16_t)(s1[i] * scale);
            }
#pragma unroll
            for (int dt = 0; dt < 4; ++dt) {
                v16bf vf = frag_b_colrows(vbuf + dt * 16 * MCHUNK, MCHUNK, sub, lane);
                acc[dt] = wmma_bf16(sa, vf, acc[dt]);
            }
        }
    }

    const int b = bh >> 4, h = bh & 15;
#pragma unroll
    for (int dt = 0; dt < 4; ++dt) {
        const int col = h * 64 + dt * 16 + (lane & 15);
#pragma unroll
        for (int r = 0; r < 8; ++r) {
            const int l = lt * 16 + r + ((lane >> 4) << 3);
            attn[((size_t)(b * LSEQ + l)) * HD + col] = (bf16_t)acc[dt][r];
        }
    }
}

// ---------------------------------------------------------------------------
// Phase 3: output GEMM: out = attn[BL x 1024] * Wo + bo   (f32 output)
// ---------------------------------------------------------------------------
__global__ void __launch_bounds__(256) out_gemm_kernel(const bf16_t* __restrict__ ab,
                                                       const bf16_t* __restrict__ wot,
                                                       const float*  __restrict__ bo,
                                                       float* __restrict__ out) {
    const int lane = threadIdx.x & 31;
    const int w    = (blockIdx.x * blockDim.x + threadIdx.x) >> 5;
    const int mt   = w >> 4;
    const int nt   = w & 15;
    const bf16_t* abase = ab + (size_t)mt * 16 * HD;

    v8f acc[4] = {};
    for (int k = 0; k < HD; k += 32) {
        __builtin_prefetch(abase + ((lane & 15) * HD) + k + 128, 0, 0);
        v16bf a = frag_a_row(abase, HD, k, lane);
#pragma unroll
        for (int dt = 0; dt < 4; ++dt) {
            v16bf b = frag_b_colrows(wot + (size_t)(nt * 64 + dt * 16) * HD, HD, k, lane);
            acc[dt] = wmma_bf16(a, b, acc[dt]);
        }
    }
#pragma unroll
    for (int dt = 0; dt < 4; ++dt) {
        const int col = nt * 64 + dt * 16 + (lane & 15);
        const float bv = bo[col];
#pragma unroll
        for (int r = 0; r < 8; ++r) {
            const int row = mt * 16 + r + ((lane >> 4) << 3);
            out[(size_t)row * DMODEL + col] = acc[dt][r] + bv;
        }
    }
}

// ---------------------------------------------------------------------------
extern "C" void kernel_launch(void* const* d_in, const int* in_sizes, int n_in,
                              void* d_out, int out_size, void* d_ws, size_t ws_size,
                              hipStream_t stream) {
    (void)in_sizes; (void)n_in; (void)out_size; (void)ws_size;
    const float* x  = (const float*)d_in[0];
    const float* Wq = (const float*)d_in[1];
    const float* bq = (const float*)d_in[2];
    const float* Wk = (const float*)d_in[3];
    const float* bk = (const float*)d_in[4];
    const float* Wv = (const float*)d_in[5];
    const float* bv = (const float*)d_in[6];
    const float* Wo = (const float*)d_in[7];
    const float* bo = (const float*)d_in[8];

    char* ws = (char*)d_ws;
    const size_t MB = 1u << 20;
    bf16_t* xb  = (bf16_t*)(ws + 0 * MB);   // 8 MiB : x in bf16
    bf16_t* wqt = (bf16_t*)(ws + 8 * MB);   // 2 MiB : Wq^T bf16
    bf16_t* wkt = (bf16_t*)(ws + 10 * MB);
    bf16_t* wvt = (bf16_t*)(ws + 12 * MB);
    bf16_t* wot = (bf16_t*)(ws + 14 * MB);
    bf16_t* Qb  = (bf16_t*)(ws + 16 * MB);  // 8 MiB : Q [b,h,l,d]
    bf16_t* Kb  = (bf16_t*)(ws + 24 * MB);  // 8 MiB : K [b,h,l,d]
    bf16_t* Vt  = (bf16_t*)(ws + 32 * MB);  // 8 MiB : V [b,h,d,l]
    bf16_t* Ab  = (bf16_t*)(ws + 40 * MB);  // 8 MiB : attn [b,l,hd]

    // Phase 0: conversions
    cvt_bf16_kernel<<<(BL * DMODEL) / 256, 256, 0, stream>>>(x, xb, BL * DMODEL);
    dim3 tb(32, 8), tg(32, 32);
    transpose_cvt_kernel<<<tg, tb, 0, stream>>>(Wq, wqt);
    transpose_cvt_kernel<<<tg, tb, 0, stream>>>(Wk, wkt);
    transpose_cvt_kernel<<<tg, tb, 0, stream>>>(Wv, wvt);
    transpose_cvt_kernel<<<tg, tb, 0, stream>>>(Wo, wot);

    // Phase 1: QKV projections (4096 waves each, 8 waves/block)
    proj_gemm_kernel<<<512, 256, 0, stream>>>(xb, wqt, bq, Qb, 0);
    proj_gemm_kernel<<<512, 256, 0, stream>>>(xb, wkt, bk, Kb, 0);
    proj_gemm_kernel<<<512, 256, 0, stream>>>(xb, wvt, bv, Vt, 1);

    // Phase 2: fused (QK^T/8) V with async-LDS K/V staging
    attn_kernel<<<512, 256, 0, stream>>>(Qb, Kb, Vt, Ab);

    // Phase 3: output projection + bias
    out_gemm_kernel<<<512, 256, 0, stream>>>(Ab, wot, bo, (float*)d_out);
}
